// JointProxyLayer_27169963115003
// MI455X (gfx1250) — compile-verified, matrix operand order checked by reading, not run on hardware
//
#include <hip/hip_runtime.h>
#include <cstdint>
#include <cstddef>

// ---- problem constants (from reference) ----
#define Bn   16
#define Tn   4096
#define XDn  512
#define ZDn  256
#define Cn   3
#define Sn   256
#define Hn   512            // 2*ZD
#define TOPKn 150
#define EPSN  1e-12f
#define Mrows (Bn*Tn)       // 65536

typedef __attribute__((ext_vector_type(16))) __bf16 v16bf;
typedef __attribute__((ext_vector_type(8)))  __bf16 v8bf;
typedef __attribute__((ext_vector_type(8)))  float  v8f;
typedef int v4i_t __attribute__((vector_size(16)));   // matches async-LDS builtin params

union ABFrag { v16bf v; v8bf h[2]; };

// CDNA5 async global->LDS path (ASYNCcnt), if the toolchain declares it.
#if defined(__HIP_DEVICE_COMPILE__) && __has_builtin(__builtin_amdgcn_global_load_async_to_lds_b128)
#define USE_ASYNC_LDS 1
#else
#define USE_ASYNC_LDS 0
#endif

#if defined(__HIP_DEVICE_COMPILE__) && __has_builtin(__builtin_amdgcn_s_wait_asynccnt)
#define WAIT_ASYNC() __builtin_amdgcn_s_wait_asynccnt(0)
#elif USE_ASYNC_LDS
#define WAIT_ASYNC() asm volatile("s_wait_asynccnt 0x0" ::: "memory")
#else
#define WAIT_ASYNC() ((void)0)
#endif

__device__ inline float softplusf(float x) {
    return fmaxf(x, 0.f) + log1pf(expf(-fabsf(x)));
}

// ---------------------------------------------------------------------------
// f32 -> bf16 bulk convert (grid-stride)
// ---------------------------------------------------------------------------
__global__ void cvt_f32_bf16(const float* __restrict__ in, __bf16* __restrict__ out, long n) {
    long i = (long)blockIdx.x * blockDim.x + threadIdx.x;
    long stride = (long)gridDim.x * blockDim.x;
    for (; i < n; i += stride) out[i] = (__bf16)in[i];
}

// WT[n][k] = (bf16) W[k][n]   (W is K x N row-major)
__global__ void transpose_w_bf16(const float* __restrict__ W, __bf16* __restrict__ WT,
                                 int K, int N) {
    int i = blockIdx.x * blockDim.x + threadIdx.x;   // over N*K, k fastest
    if (i < N * K) {
        int n = i / K, k = i % K;
        WT[i] = (__bf16)W[(size_t)k * N + n];
    }
}

// ---------------------------------------------------------------------------
// bf16 WMMA GEMM: out = act(A[M,512] * BT[N,512]^T + bias)
// Block = 256 threads (8 waves as 4x2), tile 128x128, K-step 32.
// Double-buffered LDS staged with GLOBAL_LOAD_ASYNC_TO_LDS_B128 (ASYNCcnt),
// one s_wait_asynccnt + one barrier per K-step.
// ---------------------------------------------------------------------------
template<bool RELU, bool OUT_BF16>
__global__ __launch_bounds__(256)
void gemm512_wmma(const __bf16* __restrict__ A,   // M x 512 row-major
                  const __bf16* __restrict__ BT,  // N x 512 row-major (pre-transposed W)
                  const float*  __restrict__ bias,// N
                  __bf16* __restrict__ outBf,
                  float*  __restrict__ outF,
                  int N)
{
    constexpr int K    = 512;
    constexpr int LDT  = 40;                 // padded LDS stride (elements) -> conflict-free
    constexpr int TILE = 128 * LDT;          // elements per buffer
    __shared__ __bf16 ldsA[2 * TILE];
    __shared__ __bf16 ldsB[2 * TILE];

    const int tid  = threadIdx.x;
    const int lane = tid & 31;
    const int wave = tid >> 5;
    const int h    = lane >> 4;              // half-wave: selects K sub-chunk
    const int r16  = lane & 15;              // M row (A) / N col (B) within 16-tile
    const int wm   = wave & 3;               // 4 waves along M
    const int wn   = wave >> 2;              // 2 waves along N

    const int m0 = blockIdx.x * 128;
    const int n0 = blockIdx.y * 128;

    v8f acc[2][4] = {};

    // stage one 128x32 A-tile + 128x32 B-tile into LDS buffer p
    auto stage = [&](int kk, int p) {
        #pragma unroll
        for (int i = 0; i < 2; ++i) {
            int cid = tid * 2 + i;           // 0..511 chunks of 8 bf16
            int row = cid >> 2;              // 0..127
            int kc  = (cid & 3) * 8;         // 0,8,16,24
            const __bf16* gA = &A [(size_t)(m0 + row) * K + kk + kc];
            const __bf16* gB = &BT[(size_t)(n0 + row) * K + kk + kc];
            __bf16* lA = &ldsA[p * TILE + row * LDT + kc];
            __bf16* lB = &ldsB[p * TILE + row * LDT + kc];
#if USE_ASYNC_LDS
            __builtin_amdgcn_global_load_async_to_lds_b128(
                (__attribute__((address_space(1))) v4i_t*)(void*)gA,
                (__attribute__((address_space(3))) v4i_t*)(void*)lA, 0, 0);
            __builtin_amdgcn_global_load_async_to_lds_b128(
                (__attribute__((address_space(1))) v4i_t*)(void*)gB,
                (__attribute__((address_space(3))) v4i_t*)(void*)lB, 0, 0);
#else
            *(v8bf*)lA = *(const v8bf*)gA;
            *(v8bf*)lB = *(const v8bf*)gB;
#endif
        }
    };

    stage(0, 0);

    for (int kk = 0; kk < K; kk += 32) {
        const int p = (kk >> 5) & 1;
        WAIT_ASYNC();                        // this wave's tile-p chunks are in LDS
        __syncthreads();                     // ...and so is everyone else's

        if (kk + 32 < K) stage(kk + 32, p ^ 1);
        if (kk + 64 < K) {                   // prefetch two tiles ahead (global_prefetch_b8)
            __builtin_prefetch(&A [(size_t)(m0 + (tid >> 1)) * K + kk + 64 + (tid & 1) * 16]);
            __builtin_prefetch(&BT[(size_t)(n0 + (tid >> 1)) * K + kk + 64 + (tid & 1) * 16]);
        }

        // fragments per ISA 16-bit 16x32 layout: lane r16, K chunks [h*8..+7], [16+h*8..+7]
        ABFrag af[2], bfr[4];
        #pragma unroll
        for (int mt = 0; mt < 2; ++mt) {
            int rr = p * TILE + (wm * 32 + mt * 16 + r16) * LDT;
            af[mt].h[0] = *(const v8bf*)&ldsA[rr + h * 8];
            af[mt].h[1] = *(const v8bf*)&ldsA[rr + 16 + h * 8];
        }
        #pragma unroll
        for (int nt = 0; nt < 4; ++nt) {
            int cc = p * TILE + (wn * 64 + nt * 16 + r16) * LDT;
            bfr[nt].h[0] = *(const v8bf*)&ldsB[cc + h * 8];
            bfr[nt].h[1] = *(const v8bf*)&ldsB[cc + 16 + h * 8];
        }
        #pragma unroll
        for (int mt = 0; mt < 2; ++mt)
            #pragma unroll
            for (int nt = 0; nt < 4; ++nt)
                acc[mt][nt] = __builtin_amdgcn_wmma_f32_16x16x32_bf16(
                    false, af[mt].v, false, bfr[nt].v, (short)0, acc[mt][nt], false, false);
    }

    // epilogue: C/D layout -> VGPR r: M = r + 8*h ; N = lane%16
    #pragma unroll
    for (int nt = 0; nt < 4; ++nt) {
        int col = n0 + wn * 64 + nt * 16 + r16;
        float bcol = bias[col];
        #pragma unroll
        for (int mt = 0; mt < 2; ++mt) {
            #pragma unroll
            for (int rr = 0; rr < 8; ++rr) {
                int row = m0 + wm * 32 + mt * 16 + rr + h * 8;
                float v = acc[mt][nt][rr] + bcol;
                if (RELU) v = fmaxf(v, 0.f);
                if (OUT_BF16) outBf[(size_t)row * N + col] = (__bf16)v;
                else          outF [(size_t)row * N + col] = v;
            }
        }
    }
}

// ---------------------------------------------------------------------------
// u[b,d] = (sum_t z[b,t,d]) / max(||z[b,:,d]||, eps)      (one block per b)
// ---------------------------------------------------------------------------
__global__ void colnorm_u(const float* __restrict__ z, float* __restrict__ u) {
    int b = blockIdx.x, d = threadIdx.x;           // 256 threads = d
    const float* zb = z + (size_t)b * Tn * ZDn;
    float s = 0.f, q = 0.f;
    for (int t = 0; t < Tn; ++t) {
        float v = zb[(size_t)t * ZDn + d];
        s += v; q = fmaf(v, v, q);
    }
    u[b * ZDn + d] = s / fmaxf(sqrtf(q), EPSN);
}

// ---------------------------------------------------------------------------
// per-(c,d): mu, sigma=softplus, invd = 1/max(||mu+sigma*eps[c,:,d]||, eps)
// ---------------------------------------------------------------------------
__global__ void proxy_prep(const float* __restrict__ proxies, const float* __restrict__ eps,
                           float* __restrict__ mu_s, float* __restrict__ sg_s,
                           float* __restrict__ invd) {
    int c = blockIdx.x, d = threadIdx.x;           // 3 blocks x 256 threads
    float m  = proxies[c * (2 * ZDn) + d];
    float sg = softplusf(proxies[c * (2 * ZDn) + ZDn + d]);
    float q = 0.f;
    for (int s = 0; s < Sn; ++s) {
        float v = fmaf(sg, eps[((size_t)c * Sn + s) * ZDn + d], m);
        q = fmaf(v, v, q);
    }
    mu_s[c * ZDn + d] = m;
    sg_s[c * ZDn + d] = sg;
    invd[c * ZDn + d] = 1.f / fmaxf(sqrtf(q), EPSN);
}

// att[b,c,s] = (1/T) * sum_d u[b,d] * (mu+sg*eps)[c,s,d] * invd[c,d]
__global__ void att_kernel(const float* __restrict__ u, const float* __restrict__ mu_s,
                           const float* __restrict__ sg_s, const float* __restrict__ invd,
                           const float* __restrict__ eps, float* __restrict__ att) {
    int i = blockIdx.x * blockDim.x + threadIdx.x;
    if (i >= Bn * Cn * Sn) return;
    int b = i / (Cn * Sn);
    int r = i % (Cn * Sn);
    int c = r / Sn, s = r % Sn;
    const float* ep = eps + ((size_t)c * Sn + s) * ZDn;
    float a = 0.f;
    for (int d = 0; d < ZDn; ++d) {
        float zp = fmaf(sg_s[c * ZDn + d], ep[d], mu_s[c * ZDn + d]) * invd[c * ZDn + d];
        a = fmaf(u[b * ZDn + d], zp, a);
    }
    att[i] = a * (1.f / (float)Tn);
}

// ---------------------------------------------------------------------------
// per-b top-150 means + loss term (one 256-thread block per b)
// ---------------------------------------------------------------------------
__global__ void topk_loss(const float* __restrict__ att, const int* __restrict__ y,
                          float* __restrict__ lossterm) {
    __shared__ float sPos[Sn];
    __shared__ float sNeg[Cn * Sn];
    __shared__ float red[256];
    __shared__ int   redi[256];
    const int b = blockIdx.x, t = threadIdx.x;
    const int c = y[b];
    const float NEG_INF = -3.0e38f;
    const float* ab = att + (size_t)b * Cn * Sn;

    sPos[t] = ab[c * Sn + t];
    #pragma unroll
    for (int j = 0; j < Cn; ++j) {
        int idx = t + j * 256;
        sNeg[idx] = (j == c) ? NEG_INF : ab[idx];
    }
    __syncthreads();

    float posAcc = 0.f, negAcc = 0.f;

    for (int it = 0; it < TOPKn; ++it) {          // pos: top-150 of 256
        red[t] = sPos[t]; __syncthreads();
        for (int o = 128; o > 0; o >>= 1) { if (t < o) red[t] = fmaxf(red[t], red[t + o]); __syncthreads(); }
        float m = red[0];
        redi[t] = (sPos[t] == m) ? t : 0x7fffffff; __syncthreads();
        for (int o = 128; o > 0; o >>= 1) { if (t < o) redi[t] = min(redi[t], redi[t + o]); __syncthreads(); }
        if (t == 0) { sPos[redi[0]] = NEG_INF; }
        posAcc += m;
        __syncthreads();
    }
    for (int it = 0; it < TOPKn; ++it) {          // neg: top-150 of 768
        float lm = fmaxf(fmaxf(sNeg[t], sNeg[t + 256]), sNeg[t + 512]);
        red[t] = lm; __syncthreads();
        for (int o = 128; o > 0; o >>= 1) { if (t < o) red[t] = fmaxf(red[t], red[t + o]); __syncthreads(); }
        float m = red[0];
        int li = 0x7fffffff;
        #pragma unroll
        for (int j = 0; j < Cn; ++j) { int idx = t + j * 256; if (sNeg[idx] == m) li = min(li, idx); }
        redi[t] = li; __syncthreads();
        for (int o = 128; o > 0; o >>= 1) { if (t < o) redi[t] = min(redi[t], redi[t + o]); __syncthreads(); }
        if (t == 0) { sNeg[redi[0]] = NEG_INF; }
        negAcc += m;
        __syncthreads();
    }
    if (t == 0)
        lossterm[b] = expf(negAcc / (float)TOPKn - posAcc / (float)TOPKn);
}

__global__ void finalize_loss(const float* __restrict__ lossterm, float* __restrict__ out) {
    if (threadIdx.x == 0) {
        float s = 0.f;
        for (int b = 0; b < Bn; ++b) s += lossterm[b];
        out[0] = s / (float)Bn;
    }
}

// mu_topk / sigma_topk broadcast outputs
__global__ void musig_out(const float* __restrict__ proxies,
                          float* __restrict__ out_mu, float* __restrict__ out_sg) {
    int i = blockIdx.x * blockDim.x + threadIdx.x;   // over C*ZD = 768
    if (i >= Cn * ZDn) return;
    int c = i / ZDn, d = i % ZDn;
    float m  = proxies[c * (2 * ZDn) + d];
    float sg = softplusf(proxies[c * (2 * ZDn) + ZDn + d]);
    for (int b = 0; b < Bn; ++b) {
        out_mu[(size_t)b * Cn * ZDn + i] = m;
        out_sg[(size_t)b * Cn * ZDn + i] = sg;
    }
}

// ---------------------------------------------------------------------------
extern "C" void kernel_launch(void* const* d_in, const int* in_sizes, int n_in,
                              void* d_out, int out_size, void* d_ws, size_t ws_size,
                              hipStream_t stream) {
    const float* x       = (const float*)d_in[0];
    const int*   y       = (const int*)  d_in[1];
    const float* W1      = (const float*)d_in[2];
    const float* b1      = (const float*)d_in[3];
    const float* W2      = (const float*)d_in[4];
    const float* b2      = (const float*)d_in[5];
    const float* W3      = (const float*)d_in[6];
    const float* b3      = (const float*)d_in[7];
    const float* proxies = (const float*)d_in[8];
    const float* eps     = (const float*)d_in[9];

    float* out = (float*)d_out;
    float* out_mu   = out;                       // B*C*ZD
    float* out_sg   = out + Bn * Cn * ZDn;       // B*C*ZD
    float* out_loss = out + 2 * Bn * Cn * ZDn;   // 1
    float* out_z    = out_loss + 1;              // B*T*ZD

    // workspace carve-out
    char* ws = (char*)d_ws;
    size_t off = 0;
    auto carve = [&](size_t bytes) { char* p = ws + off; off += (bytes + 255) & ~(size_t)255; return p; };
    __bf16* x_bf  = (__bf16*)carve((size_t)Mrows * XDn * 2);
    __bf16* z1_bf = (__bf16*)carve((size_t)Mrows * Hn  * 2);
    __bf16* z2_bf = (__bf16*)carve((size_t)Mrows * Hn  * 2);
    __bf16* W1T   = (__bf16*)carve((size_t)Hn  * XDn * 2);
    __bf16* W2T   = (__bf16*)carve((size_t)Hn  * Hn  * 2);
    __bf16* W3T   = (__bf16*)carve((size_t)ZDn * Hn  * 2);
    float*  u     = (float*) carve((size_t)Bn * ZDn * 4);
    float*  mu_s  = (float*) carve((size_t)Cn * ZDn * 4);
    float*  sg_s  = (float*) carve((size_t)Cn * ZDn * 4);
    float*  invd  = (float*) carve((size_t)Cn * ZDn * 4);
    float*  att   = (float*) carve((size_t)Bn * Cn * Sn * 4);
    float*  lterm = (float*) carve((size_t)Bn * 4);
    (void)ws_size; (void)in_sizes; (void)n_in; (void)out_size;

    // 1) precision prep: x -> bf16, W -> bf16 transposed
    cvt_f32_bf16<<<4096, 256, 0, stream>>>(x, x_bf, (long)Mrows * XDn);
    transpose_w_bf16<<<(Hn * XDn + 255) / 256, 256, 0, stream>>>(W1, W1T, XDn, Hn);
    transpose_w_bf16<<<(Hn * Hn  + 255) / 256, 256, 0, stream>>>(W2, W2T, Hn, Hn);
    transpose_w_bf16<<<(ZDn * Hn + 255) / 256, 256, 0, stream>>>(W3, W3T, Hn, ZDn);

    // 2) MLP via bf16 WMMA (K=512 everywhere)
    dim3 blk(256);
    gemm512_wmma<true,  true ><<<dim3(Mrows / 128, Hn  / 128), blk, 0, stream>>>(
        x_bf,  W1T, b1, z1_bf, nullptr, Hn);
    gemm512_wmma<true,  true ><<<dim3(Mrows / 128, Hn  / 128), blk, 0, stream>>>(
        z1_bf, W2T, b2, z2_bf, nullptr, Hn);
    gemm512_wmma<false, false><<<dim3(Mrows / 128, ZDn / 128), blk, 0, stream>>>(
        z2_bf, W3T, b3, nullptr, out_z, ZDn);

    // 3) factorized attention: u over T, proxy norms over S, tiny contraction
    colnorm_u<<<Bn, 256, 0, stream>>>(out_z, u);
    proxy_prep<<<Cn, 256, 0, stream>>>(proxies, eps, mu_s, sg_s, invd);
    att_kernel<<<(Bn * Cn * Sn + 255) / 256, 256, 0, stream>>>(u, mu_s, sg_s, invd, eps, att);

    // 4) top-k means + loss
    topk_loss<<<Bn, 256, 0, stream>>>(att, y, lterm);
    finalize_loss<<<1, 32, 0, stream>>>(lterm, out_loss);

    // 5) broadcast mu/sigma outputs
    musig_out<<<(Cn * ZDn + 255) / 256, 256, 0, stream>>>(proxies, out_mu, out_sg);
}
